// _SDPAAttention_66855460929593
// MI455X (gfx1250) — matrix operand order repaired and checked by reading, not compiled, not run on hardware
//
#include <hip/hip_runtime.h>
#include <hip/hip_bf16.h>
#include <math.h>

// ---------------------------------------------------------------- constants
#define B_  2
#define T_  2048
#define C_  2048
#define H_  16
#define D_  128
#define BT_ (B_ * T_)

typedef __attribute__((ext_vector_type(16))) __bf16 v16bf;
typedef __attribute__((ext_vector_type(8)))  __bf16 v8bf;
typedef __attribute__((ext_vector_type(8)))  float  v8f;

union AFragU { v16bf v; v8bf h[2]; };

__device__ __forceinline__ unsigned short f2bf(float f) {
    unsigned int u = __float_as_uint(f);
    unsigned int r = u + 0x7FFFu + ((u >> 16) & 1u);
    return (unsigned short)(r >> 16);
}

// ---- CDNA5 async global->LDS (ASYNCcnt) -------------------------------
__device__ __forceinline__ unsigned lds_off(const void* p) {
    // generic pointers to LDS carry the wave-relative byte offset in [31:0]
    return (unsigned)(unsigned long long)p;
}
__device__ __forceinline__ void async_b64(unsigned lds, unsigned long long base,
                                          unsigned voff) {
    asm volatile("global_load_async_to_lds_b64 %0, %1, %2"
                 :: "v"(lds), "v"(voff), "s"(base) : "memory");
}
__device__ __forceinline__ void async_b128(unsigned lds, unsigned long long base,
                                           unsigned voff) {
    asm volatile("global_load_async_to_lds_b128 %0, %1, %2"
                 :: "v"(lds), "v"(voff), "s"(base) : "memory");
}
__device__ __forceinline__ void wait_async0() {
    asm volatile("s_wait_asynccnt 0x0" ::: "memory");
}

// A-matrix fragment 16x32 (MxK), bf16. base points at tile origin, ld in elems.
// lane<16: M=lane, K={0..7, 16..23}; lane>=16: M=lane-16, K={8..15, 24..31}
__device__ __forceinline__ v16bf load_a_frag(const unsigned short* base, int ld, int lane) {
    int half = lane >> 4;
    int m    = lane & 15;
    const unsigned short* rp = base + (size_t)m * ld;
    AFragU f;
    f.h[0] = *(const v8bf*)(rp + half * 8);
    f.h[1] = *(const v8bf*)(rp + 16 + half * 8);
    return f.v;
}

// B-matrix fragment 32x16 (KxN), bf16, stored column-per-row: base[n*ld + k].
// lane<16: N=lane, K={0..7, 16..23}; lane>=16: N=lane-16, K={8..15, 24..31}
__device__ __forceinline__ v16bf load_b_frag(const unsigned short* base, int ld, int lane) {
    int half = lane >> 4;
    int n    = lane & 15;
    const unsigned short* rp = base + (size_t)n * ld;
    AFragU f;
    f.h[0] = *(const v8bf*)(rp + half * 8);
    f.h[1] = *(const v8bf*)(rp + 16 + half * 8);
    return f.v;
}

__device__ __forceinline__ v8f wmma_bf16(v16bf a, v16bf b, v8f c) {
    return __builtin_amdgcn_wmma_f32_16x16x32_bf16(false, a, false, b, (short)0, c,
                                                   false, false);
}

__device__ __forceinline__ v8f v8f_zero() {
    v8f z = {0.f, 0.f, 0.f, 0.f, 0.f, 0.f, 0.f, 0.f};
    return z;
}

// ---------------------------------------------------------------- kernel 1: f32 -> bf16
__global__ void cvt_bf16_kernel(const float* __restrict__ src,
                                unsigned short* __restrict__ dst, size_t n) {
    size_t i = (size_t)blockIdx.x * blockDim.x + threadIdx.x;
    size_t stride = (size_t)gridDim.x * blockDim.x;
    for (; i < n; i += stride) dst[i] = f2bf(src[i]);
}

// ---------------------------------------------------------------- kernel 2: QKV + RMSnorm + RoPE
// grid (BT/32, H), block 256.  Tile: 32 rows x 128 cols (one full head).
// Double-buffered LDS staging via async global->LDS DMA.
__global__ void __launch_bounds__(256)
qkv_rope_kernel(const unsigned short* __restrict__ xb,
                const unsigned short* __restrict__ wqb,
                const unsigned short* __restrict__ wkb,
                const unsigned short* __restrict__ wvb,
                const float* __restrict__ cosp,
                const float* __restrict__ sinp,
                unsigned short* __restrict__ qO,
                unsigned short* __restrict__ kO,
                unsigned short* __restrict__ vO) {
    __shared__ unsigned short As[2][32][32];       // x tiles  [m][k]
    __shared__ unsigned short Bs[2][3][128][32];   // W tiles  [n][k] q,k,v
    __shared__ float Of[3][32][128];               // f32 results for epilogue
    __shared__ float rstd[2][32];

    const int tid  = threadIdx.x;
    const int lane = tid & 31;
    const int wave = tid >> 5;
    const int m0   = blockIdx.x * 32;
    const int h    = blockIdx.y;
    const int n0   = h * 128;

    const unsigned short* Wp[3] = {wqb, wkb, wvb};
    const unsigned long long xbase = (unsigned long long)xb;

    // per-thread staging coordinates (fixed across k-steps)
    const int ra = (tid * 4) >> 5, ca = (tid * 4) & 31;     // A: 4 elems (b64)
    const int rb = tid >> 1,       cb = (tid & 1) * 16;     // B: 16 elems (2x b128)

    auto issue = [&](int kk, int buf) {
        async_b64(lds_off(&As[buf][ra][ca]), xbase,
                  (unsigned)(((m0 + ra) * C_ + kk + ca) * 2));
        for (int m = 0; m < 3; ++m) {
            unsigned voff = (unsigned)(((n0 + rb) * C_ + kk + cb) * 2);
            unsigned long long wb = (unsigned long long)Wp[m];
            async_b128(lds_off(&Bs[buf][m][rb][cb]),     wb, voff);
            async_b128(lds_off(&Bs[buf][m][rb][cb + 8]), wb, voff + 16);
        }
    };

    v8f acc[3][2];
    for (int m = 0; m < 3; ++m)
        for (int t = 0; t < 2; ++t) acc[m][t] = v8f_zero();

    issue(0, 0);
    const int NSTEP = C_ / 32;
    for (int step = 0; step < NSTEP; ++step) {
        const int cur = step & 1;
        wait_async0();          // this wave's portion of buf[cur] has landed
        __syncthreads();        // everyone's portion has landed
        if (step + 1 < NSTEP) issue((step + 1) * 32, cur ^ 1);

        v16bf a0 = load_a_frag(&As[cur][0][0],  32, lane);
        v16bf a1 = load_a_frag(&As[cur][16][0], 32, lane);
        for (int m = 0; m < 3; ++m) {
            v16bf bf = load_b_frag(&Bs[cur][m][wave * 16][0], 32, lane);
            acc[m][0] = wmma_bf16(a0, bf, acc[m][0]);
            acc[m][1] = wmma_bf16(a1, bf, acc[m][1]);
        }
        // next iteration's top barrier protects buf[cur] before reuse
    }

    // spill accumulators to LDS (C/D layout: row = 8*(lane>=16) + i)
    for (int m = 0; m < 3; ++m)
        for (int t = 0; t < 2; ++t)
            for (int i = 0; i < 8; ++i) {
                int row = t * 16 + ((lane >> 4) << 3) + i;
                int col = wave * 16 + (lane & 15);
                Of[m][row][col] = acc[m][t][i];
            }
    __syncthreads();

    // RMS-norm factors for q (0) and k (1) over the full head dim
    if (tid < 64) {
        int which = tid >> 5, r = tid & 31;
        float ss = 0.f;
        for (int j = 0; j < 128; ++j) { float v = Of[which][r][j]; ss += v * v; }
        rstd[which][r] = rsqrtf(ss * (1.0f / 128.0f) + 1.1920929e-7f);
    }
    __syncthreads();

    // epilogue: norm + rope + bf16 store into [B,H,T,D]
    {
        int r  = tid >> 3;
        int pp = (tid & 7) * 8;
        int tg = m0 + r;
        int bI = tg / T_, tI = tg % T_;
        float rq = rstd[0][r], rk = rstd[1][r];
        size_t base = (((size_t)bI * H_ + h) * T_ + tI) * D_;
        for (int j = 0; j < 8; ++j) {
            int d0 = pp + j;
            float cv = cosp[(size_t)tI * 64 + d0];
            float sv = sinp[(size_t)tI * 64 + d0];
            float q1 = Of[0][r][d0] * rq, q2 = Of[0][r][d0 + 64] * rq;
            qO[base + d0]      = f2bf(q1 * cv + q2 * sv);
            qO[base + d0 + 64] = f2bf(-q1 * sv + q2 * cv);
            float k1 = Of[1][r][d0] * rk, k2 = Of[1][r][d0 + 64] * rk;
            kO[base + d0]      = f2bf(k1 * cv + k2 * sv);
            kO[base + d0 + 64] = f2bf(-k1 * sv + k2 * cv);
            vO[base + d0]      = f2bf(Of[2][r][d0]);
            vO[base + d0 + 64] = f2bf(Of[2][r][d0 + 64]);
        }
    }
}

// ---------------------------------------------------------------- kernel 3: causal flash attention
// grid (T/128, H, B), block 256.  Each wave owns 16 query rows end-to-end.
__global__ void __launch_bounds__(256)
flash_kernel(const unsigned short* __restrict__ qb,
             const unsigned short* __restrict__ kb,
             const unsigned short* __restrict__ vb,
             unsigned short* __restrict__ ob) {
    __shared__ unsigned short Ks[32][128];     // K tile [key][d]
    __shared__ unsigned short Vs[128][32];     // V tile transposed [d][key]
    __shared__ float          Sf[8][16][32];   // per-wave logits
    __shared__ unsigned short Ps[8][16][32];   // per-wave probs (bf16)
    __shared__ float          alpha_s[8][16];
    __shared__ float          lsum_s[8][16];

    const int tid  = threadIdx.x;
    const int lane = tid & 31;
    const int wave = tid >> 5;
    const int q0   = blockIdx.x * 128;
    const int h    = blockIdx.y;
    const int bI   = blockIdx.z;
    const size_t headbase = ((size_t)bI * H_ + h) * T_;
    const float scale = 0.08838834764831845f;  // 1/sqrt(128)

    // Q fragments straight from global (row-major [t][d], 16B-aligned reads)
    v16bf qf[4];
    const unsigned short* qrow = qb + (headbase + q0 + wave * 16) * D_;
    for (int kc = 0; kc < 4; ++kc)
        qf[kc] = load_a_frag(qrow + kc * 32, D_, lane);

    v8f oacc[8];
    for (int ct = 0; ct < 8; ++ct) oacc[ct] = v8f_zero();
    float m_run = -INFINITY, l_run = 0.f;  // meaningful on lanes 0..15

    const unsigned long long kbase = (unsigned long long)kb;
    const int rK = tid >> 3, cK = (tid & 7) * 16;   // K: 16 elems (2x b128)

    const int kb_max = (q0 + 127) >> 5;    // inclusive causal block bound
    for (int kbk = 0; kbk <= kb_max; ++kbk) {
        // K tile via async DMA; V tile transposed through VGPRs
        {
            unsigned voff = (unsigned)(((headbase + kbk * 32 + rK) * D_ + cK) * 2);
            async_b128(lds_off(&Ks[rK][cK]),     kbase, voff);
            async_b128(lds_off(&Ks[rK][cK + 8]), kbase, voff + 16);
            const unsigned short* vsrc = vb + (headbase + kbk * 32 + rK) * D_ + cK;
            for (int j = 0; j < 16; ++j) Vs[cK + j][rK] = vsrc[j];
            if (kbk < kb_max)
                __builtin_prefetch(vb + (headbase + (kbk + 1) * 32 + rK) * D_ + cK, 0, 0);
        }
        wait_async0();
        __syncthreads();

        // S = Q K^T : 4 k-steps x 2 key col-tiles
        v8f sacc[2] = {v8f_zero(), v8f_zero()};
        for (int kc = 0; kc < 4; ++kc)
            for (int nt = 0; nt < 2; ++nt) {
                v16bf bf = load_b_frag(&Ks[nt * 16][kc * 32], 128, lane);
                sacc[nt] = wmma_bf16(qf[kc], bf, sacc[nt]);
            }
        for (int nt = 0; nt < 2; ++nt)
            for (int i = 0; i < 8; ++i)
                Sf[wave][((lane >> 4) << 3) + i][nt * 16 + (lane & 15)] = sacc[nt][i];

        // online softmax: lane r handles query row r of this wave
        if (lane < 16) {
            int r  = lane;
            int qg = q0 + wave * 16 + r;
            float bmax = -INFINITY;
            for (int j = 0; j < 32; ++j) {
                int kg = kbk * 32 + j;
                float s = (kg <= qg) ? Sf[wave][r][j] * scale : -INFINITY;
                bmax = fmaxf(bmax, s);
            }
            float newm = fmaxf(m_run, bmax);
            float al   = __expf(m_run - newm);
            float psum = 0.f;
            for (int j = 0; j < 32; ++j) {
                int kg = kbk * 32 + j;
                float s = (kg <= qg) ? Sf[wave][r][j] * scale : -INFINITY;
                float p = __expf(s - newm);
                Ps[wave][r][j] = f2bf(p);
                psum += p;
            }
            l_run = l_run * al + psum;
            m_run = newm;
            alpha_s[wave][r] = al;
            lsum_s[wave][r]  = l_run;
        }

        // rescale running O, then O += P V
        for (int i = 0; i < 8; ++i) {
            float al = alpha_s[wave][((lane >> 4) << 3) + i];
            for (int ct = 0; ct < 8; ++ct) oacc[ct][i] *= al;
        }
        v16bf pf = load_a_frag(&Ps[wave][0][0], 32, lane);
        for (int ct = 0; ct < 8; ++ct) {
            v16bf vf = load_b_frag(&Vs[ct * 16][0], 32, lane);
            oacc[ct] = wmma_bf16(pf, vf, oacc[ct]);
        }
        __syncthreads();
    }

    // normalize and store into [B, T, H*D] bf16 for the output projection
    for (int i = 0; i < 8; ++i) {
        int rloc = ((lane >> 4) << 3) + i;
        float inv = 1.0f / lsum_s[wave][rloc];
        int tI = q0 + wave * 16 + rloc;
        size_t obase = ((size_t)bI * T_ + tI) * C_ + h * 128;
        for (int ct = 0; ct < 8; ++ct)
            ob[obase + ct * 16 + (lane & 15)] = f2bf(oacc[ct][i] * inv);
    }
}

// ---------------------------------------------------------------- kernel 4: out = o @ Wo^T (f32 out)
// grid (BT/32, C/128), block 256.  Double-buffered async staging.
__global__ void __launch_bounds__(256)
out_proj_kernel(const unsigned short* __restrict__ ob,
                const unsigned short* __restrict__ wob,
                float* __restrict__ out) {
    __shared__ unsigned short As[2][32][32];
    __shared__ unsigned short Bs[2][128][32];

    const int tid  = threadIdx.x;
    const int lane = tid & 31;
    const int wave = tid >> 5;
    const int m0   = blockIdx.x * 32;
    const int n0   = blockIdx.y * 128;

    const unsigned long long obase = (unsigned long long)ob;
    const unsigned long long wbase = (unsigned long long)wob;
    const int ra = (tid * 4) >> 5, ca = (tid * 4) & 31;
    const int rb = tid >> 1,       cb = (tid & 1) * 16;

    auto issue = [&](int kk, int buf) {
        async_b64(lds_off(&As[buf][ra][ca]), obase,
                  (unsigned)(((m0 + ra) * C_ + kk + ca) * 2));
        unsigned voff = (unsigned)(((n0 + rb) * C_ + kk + cb) * 2);
        async_b128(lds_off(&Bs[buf][rb][cb]),     wbase, voff);
        async_b128(lds_off(&Bs[buf][rb][cb + 8]), wbase, voff + 16);
    };

    v8f acc[2] = {v8f_zero(), v8f_zero()};

    issue(0, 0);
    const int NSTEP = C_ / 32;
    for (int step = 0; step < NSTEP; ++step) {
        const int cur = step & 1;
        wait_async0();
        __syncthreads();
        if (step + 1 < NSTEP) issue((step + 1) * 32, cur ^ 1);

        v16bf a0 = load_a_frag(&As[cur][0][0],  32, lane);
        v16bf a1 = load_a_frag(&As[cur][16][0], 32, lane);
        v16bf bf = load_b_frag(&Bs[cur][wave * 16][0], 32, lane);
        acc[0] = wmma_bf16(a0, bf, acc[0]);
        acc[1] = wmma_bf16(a1, bf, acc[1]);
    }

    for (int t = 0; t < 2; ++t)
        for (int i = 0; i < 8; ++i) {
            int row = m0 + t * 16 + ((lane >> 4) << 3) + i;
            int col = n0 + wave * 16 + (lane & 15);
            out[(size_t)row * C_ + col] = acc[t][i];
        }
}

// ---------------------------------------------------------------- host launcher
extern "C" void kernel_launch(void* const* d_in, const int* in_sizes, int n_in,
                              void* d_out, int out_size, void* d_ws, size_t ws_size,
                              hipStream_t stream) {
    const float* x    = (const float*)d_in[0];
    const float* cosp = (const float*)d_in[1];
    const float* sinp = (const float*)d_in[2];
    const float* Wq   = (const float*)d_in[3];
    const float* Wk   = (const float*)d_in[4];
    const float* Wv   = (const float*)d_in[5];
    const float* Wo   = (const float*)d_in[6];
    float* out = (float*)d_out;

    const size_t NX = (size_t)BT_ * C_;   // x / q / k / v / o element counts
    const size_t NW = (size_t)C_ * C_;

    unsigned short* ws  = (unsigned short*)d_ws;
    unsigned short* xb  = ws;
    unsigned short* wqb = xb + NX;
    unsigned short* wkb = wqb + NW;
    unsigned short* wvb = wkb + NW;
    unsigned short* wob = wvb + NW;
    unsigned short* qb  = wob + NW;
    unsigned short* kbf = qb + NX;
    unsigned short* vbf = kbf + NX;
    unsigned short* obf = vbf + NX;

    dim3 blk(256);
    cvt_bf16_kernel<<<4096, blk, 0, stream>>>(x,  xb,  NX);
    cvt_bf16_kernel<<<2048, blk, 0, stream>>>(Wq, wqb, NW);
    cvt_bf16_kernel<<<2048, blk, 0, stream>>>(Wk, wkb, NW);
    cvt_bf16_kernel<<<2048, blk, 0, stream>>>(Wv, wvb, NW);
    cvt_bf16_kernel<<<2048, blk, 0, stream>>>(Wo, wob, NW);

    qkv_rope_kernel<<<dim3(BT_ / 32, H_), blk, 0, stream>>>(
        xb, wqb, wkb, wvb, cosp, sinp, qb, kbf, vbf);

    flash_kernel<<<dim3(T_ / 128, H_, B_), blk, 0, stream>>>(qb, kbf, vbf, obf);

    out_proj_kernel<<<dim3(BT_ / 32, C_ / 128), blk, 0, stream>>>(obf, wob, out);
}
